// Attention2D_47064251629840
// MI455X (gfx1250) — compile-verified, matrix operand order checked by reading, not run on hardware
//
#include <hip/hip_runtime.h>
#include <hip/hip_bf16.h>
#include <math.h>

typedef __bf16 bf16_t;
typedef __attribute__((ext_vector_type(16))) __bf16 v16bf;
typedef __attribute__((ext_vector_type(8)))  float    v8f;
typedef __attribute__((ext_vector_type(4)))  unsigned int v4u;

#define DIM  512
#define NPIX 4096
#define NB   4

// ---------------------------------------------------------------------------
// Fragment load for v_wmma_f32_16x16x32_bf16.
// ISA A-layout (16-bit 16x32): lane holds row m = lane&15; element e holds
// K = e + (e>=8 ? 8 : 0) + (lane>=16 ? 8 : 0).  For K-contiguous row-major
// storage this is exactly two 16-byte loads per lane.  B mirrors A with
// lane = column n, so the same loader serves B stored [n][k].
// ---------------------------------------------------------------------------
__device__ __forceinline__ v16bf frag_ld(const bf16_t* base, int row0, int k0,
                                         int ldk, int lane) {
    int m   = lane & 15;
    int off = (lane & 16) >> 1;            // 0 or 8 halves
    const bf16_t* p = base + (size_t)(row0 + m) * ldk + k0 + off;
    union { v4u u[2]; v16bf v; } r;
    r.u[0] = *(const v4u*)p;               // K = off + 0..7
    r.u[1] = *(const v4u*)(p + 16);        // K = off + 16..23
    return r.v;
}

__device__ __forceinline__ v8f wmma_bf16(v16bf a, v16bf b, v8f c) {
    return __builtin_amdgcn_wmma_f32_16x16x32_bf16(
        false, a, false, b, (short)0, c, false, false);
}

// ---------------------------------------------------------------------------
// CDNA5 async global->LDS copy: 16B per lane, tracked by ASYNCcnt, no VGPR
// round trip.  LDS address = low 32 bits of the generic pointer.
// ---------------------------------------------------------------------------
__device__ __forceinline__ void async_cp16(const bf16_t* g, void* l) {
    unsigned la = (unsigned)(uintptr_t)l;
    asm volatile("global_load_async_to_lds_b128 %0, %1, off"
                 :: "v"(la), "v"(g) : "memory");
}
__device__ __forceinline__ void wait_async0()  {
    asm volatile("s_wait_asynccnt 0" ::: "memory");
}
__device__ __forceinline__ void wait_async16() {
    asm volatile("s_wait_asynccnt 16" ::: "memory");
}

// ---------------------------------------------------------------------------
// Kernel 0: cast the four 512x512 f32 weight matrices to bf16 row-major.
// ---------------------------------------------------------------------------
__global__ void wcvt_kernel(const float* __restrict__ wq, const float* __restrict__ wk,
                            const float* __restrict__ wv, const float* __restrict__ wo,
                            bf16_t* __restrict__ dst) {
    int i = blockIdx.x * 256 + threadIdx.x;          // 0 .. 4*262144-1
    int which = i >> 18;
    int r = i & 262143;
    const float* src = (which == 0) ? wq : (which == 1) ? wk : (which == 2) ? wv : wo;
    dst[i] = (bf16_t)src[r];
}

// ---------------------------------------------------------------------------
// Kernel 1: GroupNorm. One block per (batch, group); writes pixel-major
// Xn[b][p][c] bf16.
// ---------------------------------------------------------------------------
__global__ void gn_kernel(const float* __restrict__ x, const float* __restrict__ gamma,
                          const float* __restrict__ beta, bf16_t* __restrict__ xn) {
    int blk = blockIdx.x;                 // 0..127
    int b = blk >> 5, g = blk & 31;
    const float* xb = x + ((size_t)b * DIM + g * 16) * NPIX;
    int tid = threadIdx.x;
    float s = 0.f, s2 = 0.f;
    for (int k = tid; k < 16 * NPIX; k += 256) {
        float v = xb[k];
        s += v; s2 += v * v;
    }
    __shared__ float rs[256], rs2[256];
    rs[tid] = s; rs2[tid] = s2; __syncthreads();
    for (int st = 128; st > 0; st >>= 1) {
        if (tid < st) { rs[tid] += rs[tid + st]; rs2[tid] += rs2[tid + st]; }
        __syncthreads();
    }
    float mu   = rs[0] * (1.0f / 65536.0f);
    float var  = rs2[0] * (1.0f / 65536.0f) - mu * mu;
    float rsig = rsqrtf(var + 1e-6f);
    for (int k = tid; k < 16 * NPIX; k += 256) {
        int cl = k >> 12, p = k & 4095;
        int c = g * 16 + cl;
        float v = (xb[k] - mu) * rsig * gamma[c] + beta[c];
        xn[((size_t)b * NPIX + p) * DIM + c] = (bf16_t)v;
    }
}

// ---------------------------------------------------------------------------
// Kernel 2: fused 1x1 convs (q/k/v by blockIdx.y).  64x64 tile/WG, 8 waves.
// q,k stored pixel-major [b][p][o]; v stored channel-major [b][o][p].
// ---------------------------------------------------------------------------
__global__ void __launch_bounds__(256)
conv_qkv_kernel(const bf16_t* __restrict__ Wb, const bf16_t* __restrict__ Xn,
                const float* __restrict__ bq, const float* __restrict__ bk,
                const float* __restrict__ bv,
                bf16_t* __restrict__ Qd, bf16_t* __restrict__ Kd, bf16_t* __restrict__ Vd) {
    int b = blockIdx.z, which = blockIdx.y;
    int o0 = (blockIdx.x >> 6) * 64;
    int p0 = (blockIdx.x & 63) * 64;
    const bf16_t* W = Wb + (size_t)which * DIM * DIM;
    const float* bias = (which == 0) ? bq : (which == 1) ? bk : bv;
    const bf16_t* X = Xn + (size_t)b * NPIX * DIM;
    int tid = threadIdx.x, lane = tid & 31, w = tid >> 5;
    int n = lane & 15, hi8 = (lane & 16) ? 8 : 0;
    int tm = w >> 1, np = w & 1;
    int om = o0 + tm * 16;
    v8f acc0 = {}, acc1 = {};
    for (int k0 = 0; k0 < DIM; k0 += 32) {
        v16bf a  = frag_ld(W, om, k0, DIM, lane);
        v16bf b0 = frag_ld(X, p0 + np * 32, k0, DIM, lane);
        v16bf b1 = frag_ld(X, p0 + np * 32 + 16, k0, DIM, lane);
        acc0 = wmma_bf16(a, b0, acc0);
        acc1 = wmma_bf16(a, b1, acc1);
    }
    float bvv[8];
    for (int r = 0; r < 8; r++) bvv[r] = bias[om + hi8 + r];
    for (int t = 0; t < 2; t++) {
        v8f acc = t ? acc1 : acc0;
        int p = p0 + np * 32 + t * 16 + n;
        if (which < 2) {
            bf16_t* dst = ((which == 0) ? Qd : Kd) + ((size_t)b * NPIX + p) * DIM + om + hi8;
            union { bf16_t h[8]; v4u u; } pk;
            for (int r = 0; r < 8; r++) pk.h[r] = (bf16_t)(acc[r] + bvv[r]);
            *(v4u*)dst = pk.u;                       // 8 contiguous channels
        } else {
            for (int r = 0; r < 8; r++)
                Vd[((size_t)b * DIM + om + hi8 + r) * NPIX + p] = (bf16_t)(acc[r] + bvv[r]);
        }
    }
}

// ---------------------------------------------------------------------------
// Kernel 3: flash attention, 64 queries/WG, 64-key blocks, 8 waves.
// K double-buffered + V staged with async global->LDS (ASYNCcnt), so the
// 128KB/iter LDS fill overlaps the 512 WMMAs of the iteration.
// Softmax reductions parallelized 4-way across all 256 threads.
// ---------------------------------------------------------------------------
__global__ void __launch_bounds__(256)
attn_kernel(const bf16_t* __restrict__ Qd, const bf16_t* __restrict__ Kd,
            const bf16_t* __restrict__ Vd, bf16_t* __restrict__ AO) {
    int b  = blockIdx.y;
    int q0 = blockIdx.x * 64;
    const bf16_t* Qg = Qd + (size_t)b * NPIX * DIM;
    const bf16_t* Kg = Kd + (size_t)b * NPIX * DIM;
    const bf16_t* Vg = Vd + (size_t)b * DIM * NPIX;

    __shared__ bf16_t Qs[64 * DIM];          // [j][c]   64KB
    __shared__ bf16_t Ks[2 * 64 * DIM];      // [i][c]  2x64KB double buffer
    __shared__ bf16_t Vs[DIM * 64];          // [c][i]   64KB
    __shared__ float  Ssc[64 * 64];          // [i][j]   16KB
    __shared__ bf16_t Ps[64 * 64];           // [j][i]    8KB
    __shared__ float  Pm[4 * 64], Ls[4 * 64];
    __shared__ float  Mrow[64], Lrow[64], Corr[64];

    int tid = threadIdx.x, lane = tid & 31, w = tid >> 5;
    int n = lane & 15, hi8 = (lane & 16) ? 8 : 0;
    int tj  = w & 3, tia = w >> 2;           // S tiles: (tia, tj) and (tia+2, tj)
    int tjo = w & 3, ch0 = (w >> 2) * 256;   // O tiles: j-tile tjo, channels ch0..ch0+255
    const float scale = 0.044194173824159216f;   // 512^-0.5

    // prologue: async-stage Q and K block 0 (buffer 0)
    for (int idx = tid; idx < 4096; idx += 256)
        async_cp16(Qg + (size_t)q0 * DIM + idx * 8, Qs + idx * 8);
    for (int idx = tid; idx < 4096; idx += 256)
        async_cp16(Kg + idx * 8, Ks + idx * 8);
    if (tid < 64) { Mrow[tid] = -1e30f; Lrow[tid] = 0.f; }
    v8f O[16];
    v8f zero = {};
    for (int t = 0; t < 16; t++) O[t] = zero;

    for (int it = 0; it < NPIX / 64; it++) {
        int i0 = it * 64;
        const bf16_t* Kb = Ks + (it & 1) * (64 * DIM);
        wait_async0();                       // K(it) (and Q) landed
        __syncthreads();

        // issue V(it) async; consumed only after the softmax phases
        for (int idx = tid; idx < 4096; idx += 256) {
            int c = idx >> 3, sub = idx & 7;
            async_cp16(Vg + (size_t)c * NPIX + i0 + sub * 8, Vs + c * 64 + sub * 8);
        }

        // ---- S tiles: 2 per wave, 16 WMMA k-steps over 512 channels
        v8f s0 = zero, s1 = zero;
        for (int k0 = 0; k0 < DIM; k0 += 32) {
            v16bf bb = frag_ld(Qs, tj * 16, k0, DIM, lane);
            v16bf a0 = frag_ld(Kb, tia * 16, k0, DIM, lane);
            v16bf a1 = frag_ld(Kb, (tia + 2) * 16, k0, DIM, lane);
            s0 = wmma_bf16(a0, bb, s0);
            s1 = wmma_bf16(a1, bb, s1);
        }
        for (int r = 0; r < 8; r++) {
            Ssc[(tia * 16 + hi8 + r) * 64 + tj * 16 + n]       = s0[r] * scale;
            Ssc[((tia + 2) * 16 + hi8 + r) * 64 + tj * 16 + n] = s1[r] * scale;
        }
        __syncthreads();

        // prefetch next K block into the other buffer (overlaps softmax+PV)
        if (it + 1 < NPIX / 64) {
            const bf16_t* gK = Kg + (size_t)(i0 + 64) * DIM;
            bf16_t* lK = Ks + ((it + 1) & 1) * (64 * DIM);
            for (int idx = tid; idx < 4096; idx += 256)
                async_cp16(gK + idx * 8, lK + idx * 8);
        }

        // ---- softmax over keys, 4-way parallel per query column
        {   // phase A: segmented max
            int j = tid & 63, seg = tid >> 6;
            float mb = -1e30f;
            for (int i = seg * 16; i < seg * 16 + 16; i++)
                mb = fmaxf(mb, Ssc[i * 64 + j]);
            Pm[seg * 64 + j] = mb;
        }
        __syncthreads();
        if (tid < 64) {                      // phase B: combine, running max
            int j = tid;
            float mb = fmaxf(fmaxf(Pm[j], Pm[64 + j]), fmaxf(Pm[128 + j], Pm[192 + j]));
            float mold = Mrow[j];
            float mnew = fmaxf(mold, mb);
            Corr[j] = __expf(mold - mnew);
            Mrow[j] = mnew;
        }
        __syncthreads();
        {   // phase C: exp + segmented sum, write P[j][i] bf16
            int j = tid & 63, seg = tid >> 6;
            float mnew = Mrow[j];
            float l = 0.f;
            for (int i = seg * 16; i < seg * 16 + 16; i++) {
                float pv = __expf(Ssc[i * 64 + j] - mnew);
                Ps[j * 64 + i] = (bf16_t)pv;
                l += pv;
            }
            Ls[seg * 64 + j] = l;
        }
        __syncthreads();
        if (tid < 64) {                      // phase D: running sum (overlaps O rescale)
            int j = tid;
            Lrow[j] = Lrow[j] * Corr[j] + Ls[j] + Ls[64 + j] + Ls[128 + j] + Ls[192 + j];
        }

        // ---- rescale O, wait for V, accumulate PV (2 k-steps of 32)
        float cr = Corr[tjo * 16 + n];
        for (int t = 0; t < 16; t++)
            for (int r = 0; r < 8; r++) O[t][r] *= cr;
        if (it + 1 < NPIX / 64) wait_async16();   // 16 K-prefetch ops may remain
        else                    wait_async0();    // last iter: drain V fully
        __syncthreads();
        v16bf pb0 = frag_ld(Ps, tjo * 16, 0, 64, lane);
        v16bf pb1 = frag_ld(Ps, tjo * 16, 32, 64, lane);
        for (int t = 0; t < 16; t++) {
            v16bf a0 = frag_ld(Vs, ch0 + t * 16, 0, 64, lane);
            v16bf a1 = frag_ld(Vs, ch0 + t * 16, 32, 64, lane);
            O[t] = wmma_bf16(a0, pb0, O[t]);
            O[t] = wmma_bf16(a1, pb1, O[t]);
        }
        __syncthreads();
    }
    __syncthreads();                         // Lrow final + all PV done

    // ---- normalize and store pixel-major AO[b][p][c] (one b128 per tile)
    float li = 1.0f / Lrow[tjo * 16 + n];
    int p = q0 + tjo * 16 + n;
    for (int t = 0; t < 16; t++) {
        union { bf16_t h[8]; v4u u; } pk;
        for (int r = 0; r < 8; r++) pk.h[r] = (bf16_t)(O[t][r] * li);
        *(v4u*)(AO + ((size_t)b * NPIX + p) * DIM + ch0 + t * 16 + hi8) = pk.u;
    }
}

// ---------------------------------------------------------------------------
// Kernel 4: output 1x1 conv + bias + residual, scaled by 1/sqrt(2), f32 out.
// ---------------------------------------------------------------------------
__global__ void __launch_bounds__(256)
conv_out_kernel(const bf16_t* __restrict__ Wo, const bf16_t* __restrict__ AO,
                const float* __restrict__ bo, const float* __restrict__ qin,
                float* __restrict__ out) {
    int b = blockIdx.z;
    int o0 = (blockIdx.x >> 6) * 64;
    int p0 = (blockIdx.x & 63) * 64;
    const bf16_t* X = AO + (size_t)b * NPIX * DIM;
    int tid = threadIdx.x, lane = tid & 31, w = tid >> 5;
    int n = lane & 15, hi8 = (lane & 16) ? 8 : 0;
    int tm = w >> 1, np = w & 1;
    int om = o0 + tm * 16;
    v8f acc0 = {}, acc1 = {};
    for (int k0 = 0; k0 < DIM; k0 += 32) {
        v16bf a  = frag_ld(Wo, om, k0, DIM, lane);
        v16bf b0 = frag_ld(X, p0 + np * 32, k0, DIM, lane);
        v16bf b1 = frag_ld(X, p0 + np * 32 + 16, k0, DIM, lane);
        acc0 = wmma_bf16(a, b0, acc0);
        acc1 = wmma_bf16(a, b1, acc1);
    }
    const float inv_sqrt2 = 0.7071067811865476f;
    for (int t = 0; t < 2; t++) {
        v8f acc = t ? acc1 : acc0;
        int p = p0 + np * 32 + t * 16 + n;
        for (int r = 0; r < 8; r++) {
            int o = om + hi8 + r;
            size_t idx = ((size_t)b * DIM + o) * NPIX + p;
            out[idx] = (acc[r] + bo[o] + qin[idx]) * inv_sqrt2;
        }
    }
}

// ---------------------------------------------------------------------------
extern "C" void kernel_launch(void* const* d_in, const int* in_sizes, int n_in,
                              void* d_out, int out_size, void* d_ws, size_t ws_size,
                              hipStream_t stream) {
    (void)in_sizes; (void)n_in; (void)out_size; (void)ws_size;
    const float* q     = (const float*)d_in[0];
    const float* gamma = (const float*)d_in[1];
    const float* beta  = (const float*)d_in[2];
    const float* wq = (const float*)d_in[3];
    const float* bq = (const float*)d_in[4];
    const float* wk = (const float*)d_in[5];
    const float* bk = (const float*)d_in[6];
    const float* wv = (const float*)d_in[7];
    const float* bv = (const float*)d_in[8];
    const float* wo = (const float*)d_in[9];
    const float* bo = (const float*)d_in[10];
    float* out = (float*)d_out;

    // workspace layout (bf16 elements): weights | Xn | Q | K | V | AO  (~82MB)
    bf16_t* Wb = (bf16_t*)d_ws;
    size_t wsz = (size_t)4 * DIM * DIM;
    size_t tsz = (size_t)NB * NPIX * DIM;
    bf16_t* Xn = Wb + wsz;
    bf16_t* Qd = Xn + tsz;
    bf16_t* Kd = Qd + tsz;
    bf16_t* Vd = Kd + tsz;
    bf16_t* AO = Vd + tsz;

    wcvt_kernel<<<4096, 256, 0, stream>>>(wq, wk, wv, wo, Wb);
    gn_kernel<<<128, 256, 0, stream>>>(q, gamma, beta, Xn);
    conv_qkv_kernel<<<dim3(512, 3, 4), 256, 0, stream>>>(Wb, Xn, bq, bk, bv, Qd, Kd, Vd);
    attn_kernel<<<dim3(64, 4), 256, 0, stream>>>(Qd, Kd, Vd, AO);
    conv_out_kernel<<<dim3(512, 1, 4), 256, 0, stream>>>(Wb + 3 * (size_t)DIM * DIM,
                                                         AO, bo, q, out);
}
